// TransELiteral_gate_49203145342981
// MI455X (gfx1250) — compile-verified, matrix operand order checked by reading, not run on hardware
//
#include <hip/hip_runtime.h>
#include <hip/hip_bf16.h>
#include <math.h>

#define BB 32
#define EE 30000
#define DD 200
#define LL 50
#define KPAD 256
#define NT 13            // 13*16 = 208 >= 200 output cols
#define GAMMAc 9.0f
#define ALPHAc 0.3f
#define BETAc  0.7f
#define CHUNK 1000
#define SUBE  50

typedef __attribute__((ext_vector_type(16))) __bf16 v16bf;
typedef __attribute__((ext_vector_type(8)))  float  v8f;

__device__ __forceinline__ float sigmoidf_(float x) { return 1.0f / (1.0f + expf(-x)); }
__device__ __forceinline__ float logsigf_(float x) {
    // numerically stable log(sigmoid(x)) = min(x,0) - log1p(exp(-|x|))
    return fminf(x, 0.0f) - log1pf(expf(-fabsf(x)));
}

// ---------------------------------------------------------------------------
// Kernel 1: pack Wg and [W1|W2] into bf16 WMMA B-fragment layout.
// Fragment element (s, t, ks, lane, e) -> B[k][n], n = t*16 + (lane&15),
// k = ks*32 + (e<8?0:16) + (lane>>4)*8 + (e&7).  Zero padded for n>=200, k>=250.
// ---------------------------------------------------------------------------
__global__ void k_pack_w(const float* __restrict__ Wg, const float* __restrict__ W1,
                         const float* __restrict__ W2, __bf16* __restrict__ bfrag) {
    int idx = blockIdx.x * 256 + threadIdx.x;          // 0 .. 106495
    if (idx >= 2 * NT * 8 * 32 * 16) return;
    int e    = idx & 15;
    int lane = (idx >> 4) & 31;
    int ks   = (idx >> 9) & 7;
    int rest = idx >> 12;                               // s*NT + t
    int t    = rest % NT;
    int s    = rest / NT;
    int h    = lane >> 4;
    int n    = t * 16 + (lane & 15);
    int k    = ks * 32 + ((e < 8) ? 0 : 16) + h * 8 + (e & 7);
    float v = 0.0f;
    if (n < DD) {
        if (k < DD)           v = s ? W1[n * DD + k]      : Wg[n * (DD + LL) + k];
        else if (k < DD + LL) v = s ? W2[n * LL + (k - DD)] : Wg[n * (DD + LL) + k];
    }
    bfrag[idx] = (__bf16)v;
}

// ---------------------------------------------------------------------------
// Kernel 2: literal gate over ALL E entities via WMMA bf16 GEMMs.
// Block = 256 threads (8 waves) handles 16 entity rows.  A tile staged in LDS
// in WMMA A-fragment layout; each wave sweeps N-tiles with 2 accumulators
// (Wg-path and W1/W2-path), 8 K-steps each -> 16 v_wmma per tile.
// Epilogue: out = (1-g)*x_ent + g*tanh(accG + bg),  g = sigmoid(accW + gbias).
// ---------------------------------------------------------------------------
__global__ void k_gate(const float* __restrict__ emb_e, const float* __restrict__ num_lit,
                       const float* __restrict__ bg, const float* __restrict__ gbias,
                       const __bf16* __restrict__ bfrag, float* __restrict__ e2emb) {
    __shared__ __bf16 As[8 * 32 * 16];                  // 8 KB
    const int tid = threadIdx.x;
    const int e0  = blockIdx.x * 16;                    // E = 30000 = 1875*16 exactly

    // Stage A: 16 rows x 256 K (zero padded) -> fragment layout
    for (int i = tid; i < 16 * KPAD; i += 256) {
        int m = i >> 8;
        int k = i & 255;
        int row = e0 + m;
        float v = 0.0f;
        if (k < DD)           v = emb_e[(size_t)row * DD + k];
        else if (k < DD + LL) v = num_lit[(size_t)row * LL + (k - DD)];
        int ks = k >> 5;
        int kk = k & 31;
        int h  = (kk >> 3) & 1;
        int ee = (kk & 7) + ((kk >= 16) ? 8 : 0);
        int lane = h * 16 + m;
        As[(ks * 32 + lane) * 16 + ee] = (__bf16)v;
    }
    __syncthreads();

    const int wave = tid >> 5;
    const int lane = tid & 31;
    const int h    = lane >> 4;
    const int nl   = lane & 15;

    for (int t = wave; t < NT; t += 8) {
        v8f acc0 = {0.f, 0.f, 0.f, 0.f, 0.f, 0.f, 0.f, 0.f};
        v8f acc1 = {0.f, 0.f, 0.f, 0.f, 0.f, 0.f, 0.f, 0.f};
        const __bf16* b0p = bfrag + (size_t)(0 * NT + t) * 4096;
        const __bf16* b1p = bfrag + (size_t)(1 * NT + t) * 4096;
        __builtin_prefetch(b0p + lane * 16, 0, 3);
        __builtin_prefetch(b1p + lane * 16, 0, 3);
        #pragma unroll
        for (int ks = 0; ks < 8; ++ks) {
            v16bf a  = *(const v16bf*)&As[(ks * 32 + lane) * 16];
            v16bf b0 = *(const v16bf*)(b0p + (ks * 32 + lane) * 16);
            v16bf b1 = *(const v16bf*)(b1p + (ks * 32 + lane) * 16);
            acc0 = __builtin_amdgcn_wmma_f32_16x16x32_bf16(false, a, false, b0,
                                                           (short)0, acc0, false, false);
            acc1 = __builtin_amdgcn_wmma_f32_16x16x32_bf16(false, a, false, b1,
                                                           (short)0, acc1, false, false);
        }
        int n = t * 16 + nl;
        if (n < DD) {
            float bgn = bg[n], gbn = gbias[n];
            #pragma unroll
            for (int r = 0; r < 8; ++r) {
                int row = e0 + h * 8 + r;
                float xe   = emb_e[(size_t)row * DD + n];
                float gemb = tanhf(acc0[r] + bgn);
                float g    = sigmoidf_(acc1[r] + gbn);
                e2emb[(size_t)row * DD + n] = (1.0f - g) * xe + g * gemb;
            }
        }
    }
}

// ---------------------------------------------------------------------------
// Kernel 3: q = e2_emb[e1] + emb_rel[rel]   (e1_emb is a row-gather of e2_emb)
// ---------------------------------------------------------------------------
__global__ void k_q(const int* __restrict__ e1, const int* __restrict__ rel,
                    const float* __restrict__ e2emb, const float* __restrict__ emb_rel,
                    float* __restrict__ q) {
    int i = blockIdx.x * 256 + threadIdx.x;
    if (i >= BB * DD) return;
    int b = i / DD, d = i % DD;
    q[i] = e2emb[(size_t)e1[b] * DD + d] + emb_rel[(size_t)rel[b] * DD + d];
}

// ---------------------------------------------------------------------------
// Kernel 4: pred[b][e] = sigmoid(GAMMA - sum_d |q[b][d] - e2_emb[e][d]|)
// q tile staged into LDS with GLOBAL_LOAD_ASYNC_TO_LDS_B32 (ASYNCcnt path,
// no VGPR round trip); one thread per entity then streams its e2_emb row.
// ---------------------------------------------------------------------------
__global__ void k_score(const float* __restrict__ e2emb, const float* __restrict__ q,
                        float* __restrict__ pred) {
    __shared__ float qs[BB * DD];                       // 25.6 KB
    int tid = threadIdx.x;
    // Async copy global->LDS: per-lane LDS byte address (low 32 bits of the
    // flat pointer, per the LDS-aperture rule) + per-lane 64-bit global addr.
    {
        unsigned qbase = (unsigned)(size_t)(&qs[0]);
        for (int i = tid; i < BB * DD; i += 256) {
            unsigned     ldsa = qbase + (unsigned)(i * 4);
            const float* ga   = q + i;
            asm volatile("global_load_async_to_lds_b32 %0, %1, off"
                         :: "v"(ldsa), "v"(ga) : "memory");
        }
        asm volatile("s_wait_asynccnt 0x0" ::: "memory");
    }
    __syncthreads();
    int e = blockIdx.x * 256 + tid;
    if (e >= EE) return;
    float acc[BB];
    #pragma unroll
    for (int b = 0; b < BB; ++b) acc[b] = 0.0f;
    const float* row = e2emb + (size_t)e * DD;
    for (int d = 0; d < DD; d += 4) {
        float4 v = *(const float4*)(row + d);
        #pragma unroll
        for (int b = 0; b < BB; ++b) {
            const float* qb = qs + b * DD + d;
            acc[b] += fabsf(qb[0] - v.x) + fabsf(qb[1] - v.y) +
                      fabsf(qb[2] - v.z) + fabsf(qb[3] - v.w);
        }
    }
    #pragma unroll
    for (int b = 0; b < BB; ++b)
        pred[(size_t)b * EE + e] = sigmoidf_(GAMMAc - acc[b]);
}

// ---------------------------------------------------------------------------
// Kernel 5: chunked partial sums for pos/neg means + denominators.
// Block c handles entities [c*1000, (c+1)*1000); threads 0..199 own d.
// Deterministic: per-block partials to workspace, no atomics.
// ---------------------------------------------------------------------------
__global__ void k_posneg(const int* __restrict__ e2m, const float* __restrict__ prob,
                         const float* __restrict__ drop, const float* __restrict__ e2emb,
                         float* __restrict__ part) {
    __shared__ float wpos[SUBE * 32];
    __shared__ float wneg[SUBE * 32];
    int tid = threadIdx.x;
    int c = blockIdx.x;
    int ebase = c * CHUNK;
    float accP[BB], accN[BB];
    #pragma unroll
    for (int b = 0; b < BB; ++b) { accP[b] = 0.f; accN[b] = 0.f; }
    float sposL = 0.f, sinvL = 0.f;                     // this thread's b = tid&31
    for (int eb = 0; eb < CHUNK; eb += SUBE) {
        for (int j = tid; j < SUBE * 32; j += 256) {
            int b  = j & 31;                            // == tid&31 (stride 256)
            int ee = j >> 5;
            size_t gi = (size_t)b * EE + (ebase + eb + ee);
            float m  = (float)e2m[gi];
            float wp = m * prob[gi];
            float wn = (1.0f - m) * drop[gi] * 2.0f;
            wpos[ee * 32 + b] = wp;
            wneg[ee * 32 + b] = wn;
            sposL += wp;
            sinvL += (1.0f - m);
        }
        __syncthreads();
        if (tid < DD) {
            for (int ee = 0; ee < SUBE; ++ee) {
                float v = e2emb[(size_t)(ebase + eb + ee) * DD + tid];
                #pragma unroll
                for (int b = 0; b < BB; ++b) {
                    accP[b] += wpos[ee * 32 + b] * v;
                    accN[b] += wneg[ee * 32 + b] * v;
                }
            }
        }
        __syncthreads();
    }
    float* base = part + (size_t)c * (2 * BB * DD + 2 * BB);
    if (tid < DD) {
        #pragma unroll
        for (int b = 0; b < BB; ++b) {
            base[b * DD + tid]           = accP[b];
            base[BB * DD + b * DD + tid] = accN[b];
        }
    }
    __syncthreads();
    wpos[tid] = sposL;
    wneg[tid] = sinvL;
    __syncthreads();
    if (tid < 32) {
        float sp = 0.f, si = 0.f;
        #pragma unroll
        for (int g = 0; g < 8; ++g) { sp += wpos[g * 32 + tid]; si += wneg[g * 32 + tid]; }
        base[2 * BB * DD + tid]      = sp;
        base[2 * BB * DD + BB + tid] = si;
    }
}

// ---------------------------------------------------------------------------
// Kernel 6: reduce chunks -> pos/neg means -> pos_s/neg_s -> loss.
// loss = -mean_{i,j} logsig(neg_s[j] - pos_s[i])   (faithful [B,B] broadcast)
// ---------------------------------------------------------------------------
__global__ void k_final(const int* __restrict__ e1, const float* __restrict__ q,
                        const float* __restrict__ e2emb, const float* __restrict__ part,
                        int nchunk, float* __restrict__ lossOut) {
    __shared__ float sp8[BB * 8], sn8[BB * 8];
    __shared__ float posS[BB], negS[BB], lrow[BB];
    int tid = threadIdx.x;
    int b = tid >> 3;
    int g = tid & 7;
    float pden = 0.f, iden = 0.f;
    for (int c = 0; c < nchunk; ++c) {
        const float* base = part + (size_t)c * (2 * BB * DD + 2 * BB);
        pden += base[2 * BB * DD + b];
        iden += base[2 * BB * DD + BB + b];
    }
    int e1b = e1[b];
    float psum = 0.f, nsum = 0.f;
    for (int dd = 0; dd < 25; ++dd) {
        int d = g * 25 + dd;
        float pr = 0.f, nr = 0.f;
        for (int c = 0; c < nchunk; ++c) {
            const float* base = part + (size_t)c * (2 * BB * DD + 2 * BB);
            pr += base[b * DD + d];
            nr += base[BB * DD + b * DD + d];
        }
        float e1e = e2emb[(size_t)e1b * DD + d];
        float pm = ALPHAc * (pr / pden) + (1.0f - ALPHAc) * e1e;
        float nm = BETAc  * (nr / iden) + (1.0f - BETAc)  * e1e;
        float qv = q[b * DD + d];
        psum += fabsf(qv - pm);
        nsum += fabsf(qv - nm);
    }
    sp8[b * 8 + g] = psum;
    sn8[b * 8 + g] = nsum;
    __syncthreads();
    if (g == 0) {
        float ps = 0.f, ns = 0.f;
        for (int k = 0; k < 8; ++k) { ps += sp8[b * 8 + k]; ns += sn8[b * 8 + k]; }
        posS[b] = ps;
        negS[b] = ns;
    }
    __syncthreads();
    if (tid < BB) {
        float a = 0.f;
        for (int i = 0; i < BB; ++i) a += logsigf_(negS[tid] - posS[i]);
        lrow[tid] = a;
    }
    __syncthreads();
    if (tid == 0) {
        float tot = 0.f;
        for (int j = 0; j < BB; ++j) tot += lrow[j];
        lossOut[0] = -tot / (float)(BB * BB);
    }
}

// ---------------------------------------------------------------------------
// Workspace layout (bytes):
//   [0, 24,000,000)            e2_emb  (E*D f32)
//   [24,000,000, 24,212,992)   packed bf16 B-fragments (2*13*8*32*16)
//   [24,212,992, 24,238,592)   q (B*D f32)
//   [24,238,592, ~25.8MB)      30 chunk partials of (2*B*D + 2*B) f32
// ---------------------------------------------------------------------------
extern "C" void kernel_launch(void* const* d_in, const int* in_sizes, int n_in,
                              void* d_out, int out_size, void* d_ws, size_t ws_size,
                              hipStream_t stream) {
    const int*   e1      = (const int*)  d_in[0];
    const int*   rel     = (const int*)  d_in[1];
    const int*   e2m     = (const int*)  d_in[2];
    const float* emb_e   = (const float*)d_in[3];
    const float* emb_rel = (const float*)d_in[4];
    const float* num_lit = (const float*)d_in[5];
    const float* Wg      = (const float*)d_in[6];
    const float* bg      = (const float*)d_in[7];
    const float* W1      = (const float*)d_in[8];
    const float* W2      = (const float*)d_in[9];
    const float* gbias   = (const float*)d_in[10];
    const float* prob    = (const float*)d_in[11];
    const float* drop    = (const float*)d_in[12];

    char*   ws    = (char*)d_ws;
    float*  e2emb = (float*)(ws + 0);
    __bf16* bfrag = (__bf16*)(ws + 24000000);
    float*  q     = (float*)(ws + 24212992);
    float*  part  = (float*)(ws + 24238592);

    float* pred = (float*)d_out;
    float* loss = pred + (size_t)BB * EE;

    k_pack_w<<<416, 256, 0, stream>>>(Wg, W1, W2, bfrag);
    k_gate  <<<EE / 16, 256, 0, stream>>>(emb_e, num_lit, bg, gbias, bfrag, e2emb);
    k_q     <<<(BB * DD + 255) / 256, 256, 0, stream>>>(e1, rel, e2emb, emb_rel, q);
    k_score <<<(EE + 255) / 256, 256, 0, stream>>>(e2emb, q, pred);
    k_posneg<<<EE / CHUNK, 256, 0, stream>>>(e2m, prob, drop, e2emb, part);
    k_final <<<1, 256, 0, stream>>>(e1, q, e2emb, part, EE / CHUNK, loss);
}
